// PhononPredictor_82446192214251
// MI455X (gfx1250) — compile-verified, matrix operand order checked by reading, not run on hardware
//
#include <hip/hip_runtime.h>

// ---------------------------------------------------------------------------
// PhononPredictor GNN for MI455X (gfx1250, wave32, WMMA bf16 -> f32,
// async global->LDS gather on the edge-message path, software-pipelined
// fragment loads under the WMMAs)
// ---------------------------------------------------------------------------

typedef __attribute__((ext_vector_type(16))) __bf16 v16bf;
typedef __attribute__((ext_vector_type(8)))  float  v8f;
typedef __attribute__((ext_vector_type(4)))  unsigned int uint4v;

union FragBF { v16bf v; uint4v u[2]; };

static constexpr int N_NODES = 50000;
static constexpr int N_EDGES = 800000;
static constexpr int N_GRAPH = 128;
static constexpr int H       = 128;
static constexpr int ED      = 64;
static constexpr int ND      = 92;
static constexpr int NDP     = 96;     // ND padded to multiple of 32
static constexpr int OUTC    = 30720;  // 120*256

__device__ __forceinline__ unsigned short f2bf(float f) {
  unsigned int u = __float_as_uint(f);
  u += 0x7FFFu + ((u >> 16) & 1u);     // round-to-nearest-even
  return (unsigned short)(u >> 16);
}

__device__ __forceinline__ float silu_f(float x) {
  return x * (1.0f / (1.0f + __expf(-x)));
}

__device__ __forceinline__ float wave_sum(float v) {
  v += __shfl_xor(v, 16, 32);
  v += __shfl_xor(v, 8, 32);
  v += __shfl_xor(v, 4, 32);
  v += __shfl_xor(v, 2, 32);
  v += __shfl_xor(v, 1, 32);
  return v;
}

// Async copy: 16 bytes global -> LDS, tracked by ASYNCcnt (no VGPR staging).
// lds_off is the per-lane LDS byte address (low 32 bits of the shared ptr).
__device__ __forceinline__ void async_ld_b128(unsigned lds_off, const void* gptr) {
  asm volatile("global_load_async_to_lds_b128 %0, %1, off"
               :: "v"(lds_off), "v"(gptr) : "memory");
}

__device__ __forceinline__ void wait_async0() {
  asm volatile("s_wait_asynccnt 0x0" ::: "memory");
}

// Load a 16x32 bf16 fragment (A-layout; also used for B by feeding Wt rows =
// B columns). Per CDNA5 ISA: lanes 0-15 hold row (lane), K = {0..7, 16..23};
// lanes 16-31 hold row (lane-16), K = {8..15, 24..31}. Two 16B loads.
__device__ __forceinline__ v16bf load_frag(const unsigned short* base,
                                           int stride, int row0, int k0) {
  const int lane = threadIdx.x & 31;
  const int r    = lane & 15;
  const int koff = (lane & 16) ? 8 : 0;
  const unsigned short* p = base + (size_t)(row0 + r) * stride + k0 + koff;
  FragBF f;
  f.u[0] = *(const uint4v*)(p);
  f.u[1] = *(const uint4v*)(p + 16);
  return f.v;
}

__device__ __forceinline__ v8f wmma_bf16(v16bf a, v16bf b, v8f c) {
  return __builtin_amdgcn_wmma_f32_16x16x32_bf16(false, a, false, b,
                                                 (short)0, c, false, false);
}

// One K-chunk of a 16x128 output strip: 8 WMMAs with double-buffered B
// fragments so loads for tile t+2 are in flight before the WMMA of tile t.
__device__ __forceinline__ void wmma_strip(v8f acc[8], v16bf a,
                                           const unsigned short* Wt, int ldw,
                                           int n0, int k0) {
  v16bf bb[2];
  bb[0] = load_frag(Wt, ldw, n0 + 0 * 16, k0);
  bb[1] = load_frag(Wt, ldw, n0 + 1 * 16, k0);
#pragma unroll
  for (int t = 0; t < 8; ++t) {
    v16bf bcur = bb[t & 1];
    if (t + 2 < 8) bb[t & 1] = load_frag(Wt, ldw, n0 + (t + 2) * 16, k0);
    acc[t] = wmma_bf16(a, bcur, acc[t]);
  }
}

// ---------------------------------------------------------------------------
// Utility kernels
// ---------------------------------------------------------------------------

__global__ void zero_f32_kernel(float* p, long n) {
  long i = (long)blockIdx.x * blockDim.x + threadIdx.x;
  if (i < n) p[i] = 0.0f;
}

// fp32 [rows, cin] -> bf16 [rows, cout] zero-padded
__global__ void convert_pad_kernel(const float* __restrict__ in,
                                   unsigned short* __restrict__ out,
                                   int rows, int cin, int cout) {
  long i = (long)blockIdx.x * blockDim.x + threadIdx.x;
  long total = (long)rows * cout;
  if (i >= total) return;
  int r = (int)(i / cout), c = (int)(i % cout);
  out[i] = (c < cin) ? f2bf(in[(long)r * cin + c]) : (unsigned short)0;
}

// W [K, Nout] fp32 -> Wt [Nout, Kp] bf16 (transposed, K zero-padded to Kp)
__global__ void transpose_w_kernel(const float* __restrict__ W,
                                   unsigned short* __restrict__ Wt,
                                   int K, int Nout, int Kp) {
  long i = (long)blockIdx.x * blockDim.x + threadIdx.x;
  long total = (long)Nout * Kp;
  if (i >= total) return;
  int n = (int)(i / Kp), k = (int)(i % Kp);
  Wt[i] = (k < K) ? f2bf(W[(long)k * Nout + n]) : (unsigned short)0;
}

__global__ void scatter_count_kernel(const int* __restrict__ ids,
                                     float* __restrict__ cnt, long n) {
  long i = (long)blockIdx.x * blockDim.x + threadIdx.x;
  if (i >= n) return;
  __hip_atomic_fetch_add(&cnt[ids[i]], 1.0f, __ATOMIC_RELAXED,
                         __HIP_MEMORY_SCOPE_AGENT);
}

// ---------------------------------------------------------------------------
// Generic WMMA GEMM:  C[M,Ntot] = act(A[M,K](bf16) @ Wt^T + bias)
// Wt stored [Ntot, K] bf16. Block = 128 threads = 4 waves; wave owns a
// 16-row tile x 128 cols (8 WMMA col tiles). Requires M%16==0, K%32==0,
// Ntot%128==0.
// ---------------------------------------------------------------------------
template <int ACT /*0=none,1=silu*/, int OUTF, int OUTB>
__global__ __launch_bounds__(128)
void gemm_bf16_kernel(const unsigned short* __restrict__ A, int lda,
                      const unsigned short* __restrict__ Wt,
                      const float* __restrict__ bias,
                      float* __restrict__ Cf, unsigned short* __restrict__ Cb,
                      int M, int Ntot, int K) {
  const int lane = threadIdx.x & 31;
  const int wave = threadIdx.x >> 5;
  const int row0 = (blockIdx.x * 4 + wave) * 16;
  if (row0 >= M) return;
  const int n0 = blockIdx.y * 128;

  v8f acc[8];
#pragma unroll
  for (int t = 0; t < 8; ++t) acc[t] = (v8f){};

  v16bf a = load_frag(A, lda, row0, 0);
  for (int k0 = 0; k0 < K; k0 += 32) {
    v16bf anext = a;
    if (k0 + 32 < K) anext = load_frag(A, lda, row0, k0 + 32);  // issue early
    wmma_strip(acc, a, Wt, K, n0, k0);
    a = anext;
  }

  const int nIdx  = lane & 15;
  const int mBase = (lane & 16) ? 8 : 0;
#pragma unroll
  for (int t = 0; t < 8; ++t) {
    const int n = n0 + t * 16 + nIdx;
    const float bv = bias ? bias[n] : 0.0f;
#pragma unroll
    for (int v = 0; v < 8; ++v) {
      float val = acc[t][v] + bv;
      if (ACT == 1) val = silu_f(val);
      const size_t off = (size_t)(row0 + mBase + v) * Ntot + n;
      if (OUTF) Cf[off] = val;
      if (OUTB) Cb[off] = f2bf(val);
    }
  }
}

// ---------------------------------------------------------------------------
// Fused message kernel (per GNN layer):
//   64 edges/block: async-gather [h[dst] | h[src] | edge_attr] (bf16)
//   straight into LDS (GLOBAL_LOAD_ASYNC_TO_LDS_B128, ASYNCcnt), then
//   m = msg2(silu(msg1(...))) via WMMA, atomic scatter-add into agg[dst].
// ---------------------------------------------------------------------------
__global__ __launch_bounds__(128)
void msg_kernel(const unsigned short* __restrict__ h_bf,   // [N,128]
                const unsigned short* __restrict__ ea_bf,  // [E,64]
                const int* __restrict__ ei,                // [2,E] flat
                const unsigned short* __restrict__ W1t,    // [128,320]
                const float* __restrict__ b1,
                const unsigned short* __restrict__ W2t,    // [128,128]
                const float* __restrict__ b2,
                float* __restrict__ agg,                   // [N,128] f32
                int E) {
  __shared__ unsigned short Ast[64][2 * H + ED];  // 40 KB, row = 640 B
  __shared__ unsigned short Mst[64][H];           // 16 KB
  __shared__ int sdst[64];

  const int lane = threadIdx.x & 31;
  const int wave = threadIdx.x >> 5;
  const int e0   = blockIdx.x * 64 + wave * 16;

  // ---- async gather stage: lane (r,half) streams half of edge row r ----
  {
    const int r    = lane & 15;
    const int half = lane >> 4;
    const int e    = e0 + r;
    const int src  = ei[e];
    const int dst  = ei[E + e];
    if (half == 0) sdst[wave * 16 + r] = dst;
    // LDS byte address of this edge row (generic ptr low bits = LDS offset)
    const unsigned rowoff = (unsigned)(size_t)(&Ast[wave * 16 + r][0]);
    const unsigned short* pi = h_bf + (size_t)dst * H + half * 64;
    const unsigned short* pj = h_bf + (size_t)src * H + half * 64;
    const unsigned short* pa = ea_bf + (size_t)e * ED + half * 32;
#pragma unroll
    for (int c = 0; c < 8; ++c)                    // x_i -> bytes [0,256)
      async_ld_b128(rowoff + half * 128 + c * 16, pi + c * 8);
#pragma unroll
    for (int c = 0; c < 8; ++c)                    // x_j -> bytes [256,512)
      async_ld_b128(rowoff + 256 + half * 128 + c * 16, pj + c * 8);
#pragma unroll
    for (int c = 0; c < 4; ++c)                    // edge_attr -> [512,640)
      async_ld_b128(rowoff + 512 + half * 64 + c * 16, pa + c * 8);
  }
  wait_async0();
  __syncthreads();

  const int nIdx  = lane & 15;
  const int mBase = (lane & 16) ? 8 : 0;

  // ---- GEMM1: [16,320] x [320,128] + bias, SiLU -> Mst bf16 ----
  {
    v8f acc[8];
#pragma unroll
    for (int t = 0; t < 8; ++t) acc[t] = (v8f){};
    const unsigned short* Abase = &Ast[wave * 16][0];
    v16bf a = load_frag(Abase, 2 * H + ED, 0, 0);
#pragma unroll
    for (int k0 = 0; k0 < 2 * H + ED; k0 += 32) {
      v16bf anext = a;
      if (k0 + 32 < 2 * H + ED) anext = load_frag(Abase, 2 * H + ED, 0, k0 + 32);
      wmma_strip(acc, a, W1t, 2 * H + ED, 0, k0);
      a = anext;
    }
#pragma unroll
    for (int t = 0; t < 8; ++t) {
      const float bv = b1[t * 16 + nIdx];
#pragma unroll
      for (int v = 0; v < 8; ++v)
        Mst[wave * 16 + mBase + v][t * 16 + nIdx] = f2bf(silu_f(acc[t][v] + bv));
    }
  }
  __syncthreads();

  // ---- GEMM2: [16,128] x [128,128] + bias -> atomic scatter into agg ----
  {
    v8f acc[8];
#pragma unroll
    for (int t = 0; t < 8; ++t) acc[t] = (v8f){};
    const unsigned short* Mbase = &Mst[wave * 16][0];
    v16bf a = load_frag(Mbase, H, 0, 0);
#pragma unroll
    for (int k0 = 0; k0 < H; k0 += 32) {
      v16bf anext = a;
      if (k0 + 32 < H) anext = load_frag(Mbase, H, 0, k0 + 32);
      wmma_strip(acc, a, W2t, H, 0, k0);
      a = anext;
    }
#pragma unroll
    for (int t = 0; t < 8; ++t) {
      const float bv = b2[t * 16 + nIdx];
#pragma unroll
      for (int v = 0; v < 8; ++v) {
        const int d = sdst[wave * 16 + mBase + v];
        __hip_atomic_fetch_add(&agg[(size_t)d * H + t * 16 + nIdx],
                               acc[t][v] + bv, __ATOMIC_RELAXED,
                               __HIP_MEMORY_SCOPE_AGENT);
      }
    }
  }
}

// ---------------------------------------------------------------------------
// LayerNorm (wave per row), optional f32 / bf16 outputs.
// ---------------------------------------------------------------------------
__global__ __launch_bounds__(128)
void ln_kernel(const float* __restrict__ x, const float* __restrict__ g,
               const float* __restrict__ b, float* __restrict__ of,
               unsigned short* __restrict__ ob, int rows, int D) {
  const int lane = threadIdx.x & 31;
  const int row  = blockIdx.x * 4 + (threadIdx.x >> 5);
  if (row >= rows) return;
  const float* xr = x + (size_t)row * D;
  const int nv = D / 32;
  float vals[16];
  float s = 0.0f;
  for (int i = 0; i < nv; ++i) { vals[i] = xr[lane + i * 32]; s += vals[i]; }
  const float mu = wave_sum(s) / (float)D;
  float vsum = 0.0f;
  for (int i = 0; i < nv; ++i) { float d = vals[i] - mu; vsum += d * d; }
  const float inv = rsqrtf(wave_sum(vsum) / (float)D + 1e-5f);
  for (int i = 0; i < nv; ++i) {
    const int c = lane + i * 32;
    const float y = (vals[i] - mu) * inv * g[c] + b[c];
    if (of) of[(size_t)row * D + c] = y;
    if (ob) ob[(size_t)row * D + c] = f2bf(y);
  }
}

// h' = LN(upd + h); writes f32 (in-place over hf ok) and bf16
__global__ __launch_bounds__(128)
void residual_ln_kernel(const float* __restrict__ upd, float* __restrict__ hf,
                        const float* __restrict__ g, const float* __restrict__ b,
                        unsigned short* __restrict__ hbf, int rows) {
  const int lane = threadIdx.x & 31;
  const int row  = blockIdx.x * 4 + (threadIdx.x >> 5);
  if (row >= rows) return;
  float vals[4];
  float s = 0.0f;
  for (int i = 0; i < 4; ++i) {
    const size_t off = (size_t)row * H + lane + i * 32;
    vals[i] = upd[off] + hf[off];
    s += vals[i];
  }
  const float mu = wave_sum(s) / (float)H;
  float vsum = 0.0f;
  for (int i = 0; i < 4; ++i) { float d = vals[i] - mu; vsum += d * d; }
  const float inv = rsqrtf(wave_sum(vsum) / (float)H + 1e-5f);
  for (int i = 0; i < 4; ++i) {
    const int c = lane + i * 32;
    const float y = (vals[i] - mu) * inv * g[c] + b[c];
    const size_t off = (size_t)row * H + c;
    hf[off]  = y;
    hbf[off] = f2bf(y);
  }
}

// comb[n] = [h_bf[n] | bf16(agg[n]/max(cnt[n],1))]
__global__ void concat_kernel(const unsigned short* __restrict__ hbf,
                              const float* __restrict__ agg,
                              const float* __restrict__ cnt,
                              unsigned short* __restrict__ comb, int rows) {
  long i = (long)blockIdx.x * blockDim.x + threadIdx.x;
  long total = (long)rows * 256;
  if (i >= total) return;
  const int n = (int)(i >> 8), c = (int)(i & 255);
  unsigned short v;
  if (c < H) v = hbf[(size_t)n * H + c];
  else       v = f2bf(agg[(size_t)n * H + (c - H)] / fmaxf(cnt[n], 1.0f));
  comb[i] = v;
}

__global__ __launch_bounds__(128)
void pool_kernel(const float* __restrict__ hf, const int* __restrict__ batch,
                 float* __restrict__ pool) {
  const int n = blockIdx.x;
  const int c = threadIdx.x;
  const int b = batch[n];
  __hip_atomic_fetch_add(&pool[(size_t)b * H + c], hf[(size_t)n * H + c],
                         __ATOMIC_RELAXED, __HIP_MEMORY_SCOPE_AGENT);
}

__global__ void pool_fin_kernel(const float* __restrict__ pool,
                                const float* __restrict__ cntg,
                                unsigned short* __restrict__ gbf, int rows) {
  long i = (long)blockIdx.x * blockDim.x + threadIdx.x;
  if (i >= (long)rows * H) return;
  const int gidx = (int)(i >> 7);
  gbf[i] = f2bf(pool[i] / fmaxf(cntg[gidx], 1.0f));
}

// ---------------------------------------------------------------------------
// Host orchestration
// ---------------------------------------------------------------------------
static inline unsigned int cdiv(long a, long b) { return (unsigned int)((a + b - 1) / b); }

extern "C" void kernel_launch(void* const* d_in, const int* in_sizes, int n_in,
                              void* d_out, int out_size, void* d_ws, size_t ws_size,
                              hipStream_t stream) {
  (void)in_sizes; (void)n_in; (void)out_size; (void)ws_size;
  const int N = N_NODES, E = N_EDGES, G = N_GRAPH;

  // --- inputs (jax pytree flatten: dict keys sorted) ---
  const float* x      = (const float*)d_in[0];
  const int*   ei     = (const int*)d_in[1];   // [2,E]
  const float* ea     = (const float*)d_in[2];
  const int*   batch  = (const int*)d_in[3];
  const float* e_l1_b = (const float*)d_in[4];
  const float* e_l1_w = (const float*)d_in[5];
  const float* e_l2_b = (const float*)d_in[6];
  const float* e_l2_w = (const float*)d_in[7];
  const float* e_ln_b = (const float*)d_in[8];
  const float* e_ln_g = (const float*)d_in[9];
  // gnn layer L params at 10 + 10*L: ln.b, ln.g, msg1.b, msg1.w, msg2.b,
  // msg2.w, upd1.b, upd1.w, upd2.b, upd2.w
  const float* h_l1_b  = (const float*)d_in[40];
  const float* h_l1_w  = (const float*)d_in[41];
  const float* h_l2_b  = (const float*)d_in[42];
  const float* h_l2_w  = (const float*)d_in[43];
  const float* h_l3_b  = (const float*)d_in[44];
  const float* h_l3_w  = (const float*)d_in[45];
  const float* h_ln1_b = (const float*)d_in[46];
  const float* h_ln1_g = (const float*)d_in[47];
  const float* h_ln2_b = (const float*)d_in[48];
  const float* h_ln2_g = (const float*)d_in[49];

  // --- workspace carve-up ---
  size_t off = 0;
  auto alloc = [&](size_t bytes) -> char* {
    off = (off + 255) & ~(size_t)255;
    char* p = (char*)d_ws + off;
    off += bytes;
    return p;
  };
  unsigned short* xbf   = (unsigned short*)alloc((size_t)N * NDP * 2);
  unsigned short* eabf  = (unsigned short*)alloc((size_t)E * ED * 2);
  unsigned short* hbf   = (unsigned short*)alloc((size_t)N * H * 2);
  float*          hf    = (float*)alloc((size_t)N * H * 4);
  float*          t1f   = (float*)alloc((size_t)N * 256 * 4);
  unsigned short* t1b   = (unsigned short*)alloc((size_t)N * 256 * 2);
  float*          aggf  = (float*)alloc((size_t)N * H * 4);
  unsigned short* combb = (unsigned short*)alloc((size_t)N * 256 * 2);
  float*          updf  = (float*)alloc((size_t)N * H * 4);
  float*          cntn  = (float*)alloc((size_t)N * 4);
  float*          poolf = (float*)alloc((size_t)G * H * 4);
  float*          cntg  = (float*)alloc((size_t)G * 4);
  unsigned short* gbf   = (unsigned short*)alloc((size_t)G * H * 2);
  float*          z1f   = (float*)alloc((size_t)G * 256 * 4);
  unsigned short* z1b   = (unsigned short*)alloc((size_t)G * 256 * 2);
  float*          z2f   = (float*)alloc((size_t)G * 512 * 4);
  unsigned short* z2b   = (unsigned short*)alloc((size_t)G * 512 * 2);
  unsigned short* w_e1t = (unsigned short*)alloc((size_t)H * NDP * 2);
  unsigned short* w_e2t = (unsigned short*)alloc((size_t)H * H * 2);
  unsigned short* w_m1t[3]; unsigned short* w_m2t[3];
  unsigned short* w_u1t[3]; unsigned short* w_u2t[3];
  for (int L = 0; L < 3; ++L) {
    w_m1t[L] = (unsigned short*)alloc((size_t)H * 320 * 2);
    w_m2t[L] = (unsigned short*)alloc((size_t)H * H * 2);
    w_u1t[L] = (unsigned short*)alloc((size_t)256 * 256 * 2);
    w_u2t[L] = (unsigned short*)alloc((size_t)H * 256 * 2);
  }
  unsigned short* w_h1t = (unsigned short*)alloc((size_t)256 * H * 2);
  unsigned short* w_h2t = (unsigned short*)alloc((size_t)512 * 256 * 2);
  unsigned short* w_h3t = (unsigned short*)alloc((size_t)OUTC * 512 * 2);

  const int TPB = 256;

  // --- precision prep: fp32 -> bf16, transpose weights ---
  convert_pad_kernel<<<cdiv((long)N * NDP, TPB), TPB, 0, stream>>>(x, xbf, N, ND, NDP);
  convert_pad_kernel<<<cdiv((long)E * ED, TPB), TPB, 0, stream>>>(ea, eabf, E, ED, ED);
  transpose_w_kernel<<<cdiv((long)H * NDP, TPB), TPB, 0, stream>>>(e_l1_w, w_e1t, ND, H, NDP);
  transpose_w_kernel<<<cdiv((long)H * H, TPB), TPB, 0, stream>>>(e_l2_w, w_e2t, H, H, H);
  for (int L = 0; L < 3; ++L) {
    const int base = 10 + 10 * L;
    transpose_w_kernel<<<cdiv((long)H * 320, TPB), TPB, 0, stream>>>(
        (const float*)d_in[base + 3], w_m1t[L], 320, H, 320);
    transpose_w_kernel<<<cdiv((long)H * H, TPB), TPB, 0, stream>>>(
        (const float*)d_in[base + 5], w_m2t[L], H, H, H);
    transpose_w_kernel<<<cdiv((long)256 * 256, TPB), TPB, 0, stream>>>(
        (const float*)d_in[base + 7], w_u1t[L], 256, 256, 256);
    transpose_w_kernel<<<cdiv((long)H * 256, TPB), TPB, 0, stream>>>(
        (const float*)d_in[base + 9], w_u2t[L], 256, H, 256);
  }
  transpose_w_kernel<<<cdiv((long)256 * H, TPB), TPB, 0, stream>>>(h_l1_w, w_h1t, H, 256, H);
  transpose_w_kernel<<<cdiv((long)512 * 256, TPB), TPB, 0, stream>>>(h_l2_w, w_h2t, 256, 512, 256);
  transpose_w_kernel<<<cdiv((long)OUTC * 512, TPB), TPB, 0, stream>>>(h_l3_w, w_h3t, 512, OUTC, 512);

  // --- degree / graph counts (constant over layers) ---
  zero_f32_kernel<<<cdiv(N, TPB), TPB, 0, stream>>>(cntn, N);
  zero_f32_kernel<<<cdiv(G, TPB), TPB, 0, stream>>>(cntg, G);
  scatter_count_kernel<<<cdiv(E, TPB), TPB, 0, stream>>>(ei + E, cntn, E);
  scatter_count_kernel<<<cdiv(N, TPB), TPB, 0, stream>>>(batch, cntg, N);

  // --- embed: h = l2(LN(silu(l1(x)))) ---
  gemm_bf16_kernel<1, 1, 0><<<dim3(cdiv(N, 64), 1), 128, 0, stream>>>(
      xbf, NDP, w_e1t, e_l1_b, t1f, nullptr, N, H, NDP);
  ln_kernel<<<cdiv(N, 4), 128, 0, stream>>>(t1f, e_ln_g, e_ln_b, nullptr, t1b, N, H);
  gemm_bf16_kernel<0, 1, 1><<<dim3(cdiv(N, 64), 1), 128, 0, stream>>>(
      t1b, H, w_e2t, e_l2_b, hf, hbf, N, H, H);

  // --- GNN layers ---
  for (int L = 0; L < 3; ++L) {
    const int base = 10 + 10 * L;
    const float* ln_b = (const float*)d_in[base + 0];
    const float* ln_g = (const float*)d_in[base + 1];
    const float* m1b  = (const float*)d_in[base + 2];
    const float* m2b  = (const float*)d_in[base + 4];
    const float* u1b  = (const float*)d_in[base + 6];
    const float* u2b  = (const float*)d_in[base + 8];

    zero_f32_kernel<<<cdiv((long)N * H, TPB), TPB, 0, stream>>>(aggf, (long)N * H);
    msg_kernel<<<E / 64, 128, 0, stream>>>(hbf, eabf, ei, w_m1t[L], m1b,
                                           w_m2t[L], m2b, aggf, E);
    concat_kernel<<<cdiv((long)N * 256, TPB), TPB, 0, stream>>>(hbf, aggf, cntn, combb, N);
    gemm_bf16_kernel<1, 0, 1><<<dim3(cdiv(N, 64), 2), 128, 0, stream>>>(
        combb, 256, w_u1t[L], u1b, nullptr, t1b, N, 256, 256);
    gemm_bf16_kernel<0, 1, 0><<<dim3(cdiv(N, 64), 1), 128, 0, stream>>>(
        t1b, 256, w_u2t[L], u2b, updf, nullptr, N, H, 256);
    residual_ln_kernel<<<cdiv(N, 4), 128, 0, stream>>>(updf, hf, ln_g, ln_b, hbf, N);
  }

  // --- global mean pool ---
  zero_f32_kernel<<<cdiv((long)G * H, TPB), TPB, 0, stream>>>(poolf, (long)G * H);
  pool_kernel<<<N, H, 0, stream>>>(hf, batch, poolf);
  pool_fin_kernel<<<cdiv((long)G * H, TPB), TPB, 0, stream>>>(poolf, cntg, gbf, G);

  // --- head ---
  gemm_bf16_kernel<1, 1, 0><<<dim3(cdiv(G, 64), 2), 128, 0, stream>>>(
      gbf, H, w_h1t, h_l1_b, z1f, nullptr, G, 256, H);
  ln_kernel<<<cdiv(G, 4), 128, 0, stream>>>(z1f, h_ln1_g, h_ln1_b, nullptr, z1b, G, 256);
  gemm_bf16_kernel<1, 1, 0><<<dim3(cdiv(G, 64), 4), 128, 0, stream>>>(
      z1b, 256, w_h2t, h_l2_b, z2f, nullptr, G, 512, 256);
  ln_kernel<<<cdiv(G, 4), 128, 0, stream>>>(z2f, h_ln2_g, h_ln2_b, nullptr, z2b, G, 512);
  gemm_bf16_kernel<0, 1, 0><<<dim3(cdiv(G, 64), OUTC / 128), 128, 0, stream>>>(
      z2b, 512, w_h3t, h_l3_b, (float*)d_out, nullptr, G, OUTC, 512);
}
